// CausalSelfAttention_45801531244965
// MI455X (gfx1250) — compile-verified, matrix operand order checked by reading, not run on hardware
//
#include <hip/hip_runtime.h>
#include <hip/hip_bf16.h>

// ---------------------------------------------------------------------------
// CausalSelfAttention on gfx1250 (MI455X), bf16 WMMA pipeline.
//   D_MODEL=1024, N_HEADS=16, HEAD_DIM=64, B=4, T=2048.
// ---------------------------------------------------------------------------

typedef __attribute__((ext_vector_type(16))) __bf16 v16bf;
typedef __attribute__((ext_vector_type(8)))  __bf16 bf16x8;
typedef __attribute__((ext_vector_type(8)))  float  v8f;

#define D_MODEL   1024
#define N_HEADS   16
#define HEAD_DIM  64
#define BATCH     4
#define SEQ       2048
#define TOKENS    (BATCH * SEQ)        // 8192
#define QKV_N     (3 * D_MODEL)        // 3072

// WMMA helper: D = A(16x32 bf16) * B(32x16 bf16) + C(f32)
static __device__ __forceinline__ v8f wmma_bf16(v16bf a, v16bf b, v8f c) {
    return __builtin_amdgcn_wmma_f32_16x16x32_bf16(
        /*neg_a=*/false, a, /*neg_b=*/false, b,
        /*c_mod=*/(short)0, c, /*reuse_a=*/false, /*reuse_b=*/false);
}

// Load one 16x32 (A-layout) or 32x16 (B-layout) bf16 fragment for this lane.
// Per the CDNA5 16-bit operand layout, lane (h = lane>>4) needs two
// contiguous 8-element chunks: k in [8h, 8h+8) and [16+8h, 24+8h).
// `row` must point at (matrix row for A / matrix column for B) + k0.
static __device__ __forceinline__ v16bf load_frag(const __bf16* row, int h) {
    bf16x8 lo = *(const bf16x8*)(row + 8 * h);
    bf16x8 hi = *(const bf16x8*)(row + 16 + 8 * h);
    v16bf r;
#pragma unroll
    for (int i = 0; i < 8; ++i) { r[i] = lo[i]; r[i + 8] = hi[i]; }
    return r;
}

// ---------------------------------------------------------------------------
// Conversion kernels
// ---------------------------------------------------------------------------
__global__ void k_cvt_f32_bf16(const float* __restrict__ in,
                               __bf16* __restrict__ out, int n) {
    int i = blockIdx.x * blockDim.x + threadIdx.x;
    if (i < n) out[i] = (__bf16)in[i];
}

// W[K][N] fp32 (row-major) -> Wt[N][K] bf16 (row-major), i.e. transpose.
__global__ void k_cvt_transpose(const float* __restrict__ W,
                                __bf16* __restrict__ Wt, int K, int N) {
    int i = blockIdx.x * blockDim.x + threadIdx.x;   // i = n*K + k
    if (i >= K * N) return;
    int n = i / K;
    int k = i - n * K;
    Wt[i] = (__bf16)W[(size_t)k * N + n];
}

// ---------------------------------------------------------------------------
// GEMM:  out[M][N] = A[M][K](bf16) * Bt[N][K](bf16)^T + bias[N]
// Block: 256 threads = 8 waves; wave tile 32x64; block tile 128x128.
// grid = (N/128, M/128). K multiple of 32.
// ---------------------------------------------------------------------------
template <bool F32OUT>
__global__ __launch_bounds__(256)
void k_gemm_bf16(const __bf16* __restrict__ A, const __bf16* __restrict__ Bt,
                 const float* __restrict__ bias, void* __restrict__ outp,
                 int M, int N, int K) {
    const int lane = threadIdx.x & 31;
    const int w    = threadIdx.x >> 5;
    const int wm   = w & 3;          // 4 waves along M
    const int wn   = w >> 2;         // 2 waves along N
    const int ln   = lane & 15;
    const int h    = lane >> 4;

    const int rowbase = blockIdx.y * 128 + wm * 32;   // 2 m-tiles of 16
    const int colbase = blockIdx.x * 128 + wn * 64;   // 4 n-tiles of 16

    const __bf16* arow[2];
#pragma unroll
    for (int mi = 0; mi < 2; ++mi)
        arow[mi] = A + (size_t)(rowbase + mi * 16 + ln) * K;
    const __bf16* brow[4];
#pragma unroll
    for (int ni = 0; ni < 4; ++ni)
        brow[ni] = Bt + (size_t)(colbase + ni * 16 + ln) * K;

    v8f acc[2][4] = {};

    for (int k0 = 0; k0 < K; k0 += 32) {
        v16bf af[2];
#pragma unroll
        for (int mi = 0; mi < 2; ++mi) af[mi] = load_frag(arow[mi] + k0, h);
        v16bf bfv[4];
#pragma unroll
        for (int ni = 0; ni < 4; ++ni) bfv[ni] = load_frag(brow[ni] + k0, h);
#pragma unroll
        for (int mi = 0; mi < 2; ++mi)
#pragma unroll
            for (int ni = 0; ni < 4; ++ni)
                acc[mi][ni] = wmma_bf16(af[mi], bfv[ni], acc[mi][ni]);
    }

    // Epilogue: C layout -> lane holds column (colbase+nt*16+ln),
    // VGPR j holds row (rowbase+mi*16+j+8h).
#pragma unroll
    for (int ni = 0; ni < 4; ++ni) {
        const int col = colbase + ni * 16 + ln;
        const float bv = bias[col];
#pragma unroll
        for (int mi = 0; mi < 2; ++mi) {
#pragma unroll
            for (int j = 0; j < 8; ++j) {
                const int row = rowbase + mi * 16 + j + 8 * h;
                const float v = acc[mi][ni][j] + bv;
                if (F32OUT)
                    ((float*)outp)[(size_t)row * N + col] = v;
                else
                    ((__bf16*)outp)[(size_t)row * N + col] = (__bf16)v;
            }
        }
    }
}

// ---------------------------------------------------------------------------
// Flash-style causal attention.
// qkv: [TOKENS][3072] bf16 (q | k | v).  y: [TOKENS][1024] bf16.
// grid = (SEQ/128, BATCH*N_HEADS); block = 256 threads (8 waves).
// Wave w handles queries [qbase + 16w, qbase + 16w + 16).
// ---------------------------------------------------------------------------
__global__ __launch_bounds__(256)
void k_attention(const __bf16* __restrict__ qkv, __bf16* __restrict__ y) {
    __shared__ __align__(16) __bf16 Vt[HEAD_DIM][40];     // V^T tile, 32 keys
    __shared__ __align__(16) __bf16 Pb[8][16][40];        // per-wave P tile

    const int bh    = blockIdx.y;
    const int b     = bh >> 4;
    const int head  = bh & 15;
    const int qbase = blockIdx.x * 128;

    const int lane = threadIdx.x & 31;
    const int w    = threadIdx.x >> 5;
    const int ln   = lane & 15;
    const int hh   = lane >> 4;
    const int qtile = qbase + w * 16;

    const __bf16* base = qkv + (size_t)b * SEQ * QKV_N;
    const int qoff = head * HEAD_DIM;
    const int koff = D_MODEL + head * HEAD_DIM;
    const int voff = 2 * D_MODEL + head * HEAD_DIM;

    // Q fragments (A-matrix 16x32 for d-chunks {0..31} and {32..63})
    const __bf16* qrow = base + (size_t)(qtile + ln) * QKV_N + qoff;
    v16bf qf[2];
    qf[0] = load_frag(qrow + 0, hh);
    qf[1] = load_frag(qrow + 32, hh);

    float rmax[8], rsum[8];
#pragma unroll
    for (int j = 0; j < 8; ++j) { rmax[j] = -3.0e38f; rsum[j] = 0.0f; }
    v8f acc[4] = {};

    const int nkb = qbase / 32 + 4;   // key blocks covering keys <= qbase+127
    for (int kb = 0; kb < nkb; ++kb) {
        const int kbase = kb * 32;

        __syncthreads();
        // Cooperatively stage V tile transposed: Vt[d][key] for 32 keys.
        {
            const int key = threadIdx.x >> 3;           // 0..31
            const int d8  = (threadIdx.x & 7) * 8;      // 0..56
            bf16x8 vv = *(const bf16x8*)(base + (size_t)(kbase + key) * QKV_N
                                         + voff + d8);
#pragma unroll
            for (int i = 0; i < 8; ++i) Vt[d8 + i][key] = vv[i];
        }
        __syncthreads();

        // S = Q K^T for this wave's 16 queries x 32 keys (two 16x16 tiles).
        v8f s0 = {}, s1 = {};
        {
            const __bf16* krow0 = base + (size_t)(kbase + ln) * QKV_N + koff;
            const __bf16* krow1 = base + (size_t)(kbase + 16 + ln) * QKV_N + koff;
            v16bf kf;
            kf = load_frag(krow0 + 0, hh);  s0 = wmma_bf16(qf[0], kf, s0);
            kf = load_frag(krow0 + 32, hh); s0 = wmma_bf16(qf[1], kf, s0);
            kf = load_frag(krow1 + 0, hh);  s1 = wmma_bf16(qf[0], kf, s1);
            kf = load_frag(krow1 + 32, hh); s1 = wmma_bf16(qf[1], kf, s1);
        }

        // Online softmax across the 32-key block.
#pragma unroll
        for (int j = 0; j < 8; ++j) {
            const int q = qtile + j + 8 * hh;
            const float v0 = s0[j] * 0.125f;            // 1/sqrt(64)
            const float v1 = s1[j] * 0.125f;
            const bool m0 = (kbase + ln) > q;
            const bool m1 = (kbase + 16 + ln) > q;
            float tm = fmaxf(m0 ? -3.0e38f : v0, m1 ? -3.0e38f : v1);
#pragma unroll
            for (int mk = 8; mk >= 1; mk >>= 1)
                tm = fmaxf(tm, __shfl_xor(tm, mk, 32));
            const float nm = fmaxf(rmax[j], tm);
            const float f  = __expf(rmax[j] - nm);      // 0 on first block
            const float p0 = m0 ? 0.0f : __expf(v0 - nm);
            const float p1 = m1 ? 0.0f : __expf(v1 - nm);
            float ps = p0 + p1;
#pragma unroll
            for (int mk = 8; mk >= 1; mk >>= 1)
                ps += __shfl_xor(ps, mk, 32);
            rsum[j] = rsum[j] * f + ps;
            rmax[j] = nm;
#pragma unroll
            for (int nt = 0; nt < 4; ++nt) acc[nt][j] = acc[nt][j] * f;
            // C-layout -> A-layout round trip through per-wave LDS.
            Pb[w][j + 8 * hh][ln]      = (__bf16)p0;
            Pb[w][j + 8 * hh][16 + ln] = (__bf16)p1;
        }

        // Make this wave's P stores visible to its own fragment loads.
        asm volatile("s_wait_dscnt 0" ::: "memory");

        const v16bf pf = load_frag(&Pb[w][ln][0], hh);
#pragma unroll
        for (int nt = 0; nt < 4; ++nt) {
            const v16bf vf = load_frag(&Vt[nt * 16 + ln][0], hh);
            acc[nt] = wmma_bf16(pf, vf, acc[nt]);
        }
    }

    // Normalize and write y[token][head*64 + d] as bf16.
#pragma unroll
    for (int nt = 0; nt < 4; ++nt) {
#pragma unroll
        for (int j = 0; j < 8; ++j) {
            const int row = b * SEQ + qtile + j + 8 * hh;
            const int col = head * HEAD_DIM + nt * 16 + ln;
            y[(size_t)row * D_MODEL + col] = (__bf16)(acc[nt][j] / rsum[j]);
        }
    }
}

// ---------------------------------------------------------------------------
// Host-side launch
// ---------------------------------------------------------------------------
extern "C" void kernel_launch(void* const* d_in, const int* in_sizes, int n_in,
                              void* d_out, int out_size, void* d_ws, size_t ws_size,
                              hipStream_t stream) {
    (void)in_sizes; (void)n_in; (void)out_size; (void)ws_size;

    const float* x      = (const float*)d_in[0];   // [8192,1024]
    const float* W_qkv  = (const float*)d_in[1];   // [1024,3072]
    const float* b_qkv  = (const float*)d_in[2];   // [3072]
    const float* W_proj = (const float*)d_in[3];   // [1024,1024]
    const float* b_proj = (const float*)d_in[4];   // [1024]
    float* out = (float*)d_out;                    // [8192,1024]

    // Workspace carve (bf16 buffers), 256B aligned.
    size_t off = 0;
    auto carve = [&](size_t bytes) {
        void* p = (char*)d_ws + off;
        off += (bytes + 255) & ~(size_t)255;
        return p;
    };
    __bf16* xbf    = (__bf16*)carve((size_t)TOKENS * D_MODEL * 2);
    __bf16* wqkvt  = (__bf16*)carve((size_t)QKV_N  * D_MODEL * 2);
    __bf16* wprojt = (__bf16*)carve((size_t)D_MODEL * D_MODEL * 2);
    __bf16* qkv    = (__bf16*)carve((size_t)TOKENS * QKV_N * 2);
    __bf16* ybf    = (__bf16*)carve((size_t)TOKENS * D_MODEL * 2);

    // 1) Precision conversions / weight transposes.
    {
        int n = TOKENS * D_MODEL;
        k_cvt_f32_bf16<<<(n + 255) / 256, 256, 0, stream>>>(x, xbf, n);
    }
    {
        int n = D_MODEL * QKV_N;
        k_cvt_transpose<<<(n + 255) / 256, 256, 0, stream>>>(W_qkv, wqkvt,
                                                             D_MODEL, QKV_N);
    }
    {
        int n = D_MODEL * D_MODEL;
        k_cvt_transpose<<<(n + 255) / 256, 256, 0, stream>>>(W_proj, wprojt,
                                                             D_MODEL, D_MODEL);
    }

    // 2) QKV GEMM: [8192,1024] x [1024,3072] + b_qkv -> bf16 qkv.
    k_gemm_bf16<false><<<dim3(QKV_N / 128, TOKENS / 128), 256, 0, stream>>>(
        xbf, wqkvt, b_qkv, (void*)qkv, TOKENS, QKV_N, D_MODEL);

    // 3) Causal flash attention -> ybf [8192,1024] bf16.
    k_attention<<<dim3(SEQ / 128, BATCH * N_HEADS), 256, 0, stream>>>(qkv, ybf);

    // 4) Output projection: [8192,1024] x [1024,1024] + b_proj -> fp32 out.
    k_gemm_bf16<true><<<dim3(D_MODEL / 128, TOKENS / 128), 256, 0, stream>>>(
        ybf, wprojt, b_proj, (void*)out, TOKENS, D_MODEL, D_MODEL);
}
